// MultiHeadAttention_8693013807757
// MI455X (gfx1250) — matrix operand order repaired and checked by reading, compiled, not run on hardware
//
#include <hip/hip_runtime.h>
#include <hip/hip_bf16.h>
#include <math.h>

typedef __attribute__((ext_vector_type(16))) _Float16 v16h;
typedef __attribute__((ext_vector_type(8)))  _Float16 v8h;
typedef __attribute__((ext_vector_type(8)))  float    v8f;
typedef __attribute__((ext_vector_type(4)))  float    v4f;

#define BB 8
#define SS 1024
#define HH 8
#define DD 64

__device__ __forceinline__ v8f vzero8() {
  v8f z;
#pragma unroll
  for (int i = 0; i < 8; ++i) z[i] = 0.0f;
  return z;
}

// WMMA wrapper: D = A(16x32 f16) * B(32x16 f16) + C(16x16 f32)
__device__ __forceinline__ v8f wmma_f16(v16h a, v16h b, v8f c) {
  return __builtin_amdgcn_wmma_f32_16x16x32_f16(false, a, false, b, (short)0, c,
                                                false, false);
}

// A-fragment (16x32, M x K) from row-major f16 LDS [row][k], row stride ld halves.
// Layout (ISA 7.12.2): lane m = lane&15; khalf = (lane>=16)?8:0;
// halves 0..7 -> K = khalf+0..7 ; halves 8..15 -> K = 16+khalf+0..7
__device__ __forceinline__ v16h load_A_lds(const _Float16* p, int ld, int m_base,
                                           int k_base) {
  const int lane = threadIdx.x & 31;
  const int m = m_base + (lane & 15);
  const int khalf = (lane >> 4) << 3;
  const _Float16* q = p + m * ld + k_base + khalf;
  v8h lo = *(const v8h*)(q);
  v8h hi = *(const v8h*)(q + 16);
  v16h a;
#pragma unroll
  for (int i = 0; i < 8; ++i) { a[i] = lo[i]; a[i + 8] = hi[i]; }
  return a;
}

// Same A-fragment gather, but directly from global f32 (converted to f16).
__device__ __forceinline__ v16h load_A_gf32(const float* base, int ld, int k_base) {
  const int lane = threadIdx.x & 31;
  const int m = lane & 15;
  const int khalf = (lane >> 4) << 3;
  const float* q = base + (size_t)m * ld + k_base + khalf;
  v4f x0 = *(const v4f*)(q);
  v4f x1 = *(const v4f*)(q + 4);
  v4f y0 = *(const v4f*)(q + 16);
  v4f y1 = *(const v4f*)(q + 20);
  v16h a;
#pragma unroll
  for (int i = 0; i < 4; ++i) {
    a[i]      = (_Float16)x0[i];
    a[i + 4]  = (_Float16)x1[i];
    a[i + 8]  = (_Float16)y0[i];
    a[i + 12] = (_Float16)y1[i];
  }
  return a;
}

// B-fragment (32x16, K x N) from n-major f16 LDS [n][k], row stride ld halves.
// Layout: lane n = lane&15; sec = (lane>=16)?8:0; halves 0..7 -> K = sec+0..7 ; halves 8..15 -> K = 16+sec+0..7
__device__ __forceinline__ v16h load_B_lds(const _Float16* p, int ld, int n_base,
                                           int k_base) {
  const int lane = threadIdx.x & 31;
  const int n = n_base + (lane & 15);
  const int sec = (lane >> 4) << 3;
  const _Float16* q = p + n * ld + k_base + sec;
  v8h lo = *(const v8h*)(q);
  v8h hi = *(const v8h*)(q + 16);
  v16h b;
#pragma unroll
  for (int i = 0; i < 8; ++i) { b[i] = lo[i]; b[i + 8] = hi[i]; }
  return b;
}

// ---------------------------------------------------------------------------
// Kernel 1: projection GEMM  C[M,N] = A[M,K] * W[K,N] + bias   (f32 in, f16
// WMMA compute, f32 out).  64x64 tile per 128-thread block, 4 waves.
// ---------------------------------------------------------------------------
#define G_LD 72
__global__ __launch_bounds__(128) void gemm_proj(const float* __restrict__ A,
                                                 const float* __restrict__ W,
                                                 const float* __restrict__ bias,
                                                 float* __restrict__ C, int M,
                                                 int N, int K) {
  __shared__ alignas(16) _Float16 As[64 * G_LD];   // [m][k]
  __shared__ alignas(16) _Float16 Ws[64 * G_LD];   // transposed: [n][k]
  const int t = threadIdx.x;
  const int wave = t >> 5;
  const int lane = t & 31;
  const int m0 = blockIdx.x * 64;
  const int n0 = blockIdx.y * 64;

  v8f acc[4];
#pragma unroll
  for (int i = 0; i < 4; ++i) acc[i] = vzero8();

  for (int kt = 0; kt < K; kt += 64) {
    __syncthreads();
    // A tile: 64 rows x 64 k, f32 -> f16. Gather first, then convert+store.
    v4f ax[8], wx[8];
#pragma unroll
    for (int rep = 0; rep < 8; ++rep) {
      int i = rep * 128 + t;
      int r = i >> 4, c4 = (i & 15) << 2;
      ax[rep] = *(const v4f*)(A + (size_t)(m0 + r) * K + kt + c4);
      int kk = i >> 4, n4 = (i & 15) << 2;
      wx[rep] = *(const v4f*)(W + (size_t)(kt + kk) * N + n0 + n4);
    }
    // prefetch next tiles into caches while this tile is consumed
    if (kt + 64 < K) {
      int r = t >> 1, part = (t & 1) * 32;
      __builtin_prefetch(A + (size_t)(m0 + r) * K + kt + 64 + part, 0, 0);
      __builtin_prefetch(W + (size_t)(kt + 64 + r) * N + n0 + part, 0, 0);
    }
#pragma unroll
    for (int rep = 0; rep < 8; ++rep) {
      int i = rep * 128 + t;
      int r = i >> 4, c4 = (i & 15) << 2;
      _Float16* d = As + r * G_LD + c4;
      d[0] = (_Float16)ax[rep][0]; d[1] = (_Float16)ax[rep][1];
      d[2] = (_Float16)ax[rep][2]; d[3] = (_Float16)ax[rep][3];
      int kk = i >> 4, n4 = (i & 15) << 2;
      Ws[(n4 + 0) * G_LD + kk] = (_Float16)wx[rep][0];
      Ws[(n4 + 1) * G_LD + kk] = (_Float16)wx[rep][1];
      Ws[(n4 + 2) * G_LD + kk] = (_Float16)wx[rep][2];
      Ws[(n4 + 3) * G_LD + kk] = (_Float16)wx[rep][3];
    }
    __syncthreads();
#pragma unroll
    for (int c = 0; c < 2; ++c) {
      v16h a = load_A_lds(As, G_LD, wave * 16, c * 32);
#pragma unroll
      for (int nt = 0; nt < 4; ++nt) {
        v16h b = load_B_lds(Ws, G_LD, nt * 16, c * 32);
        acc[nt] = wmma_f16(a, b, acc[nt]);
      }
    }
  }
  // store (C/D layout: VGPR r -> row r + 8*lanehalf; lane&15 -> col)
  const int ncol = lane & 15;
  const int mrow0 = m0 + wave * 16 + ((lane >> 4) << 3);
#pragma unroll
  for (int nt = 0; nt < 4; ++nt) {
    int n = n0 + nt * 16 + ncol;
    float bvv = bias[n];
#pragma unroll
    for (int r = 0; r < 8; ++r)
      C[(size_t)(mrow0 + r) * N + n] = acc[nt][r] + bvv;
  }
}

// ---------------------------------------------------------------------------
// Kernel 2: differential flash attention.
// block = (b,h, q-tile of 64); 4 waves, each owns 16 q rows.
// ---------------------------------------------------------------------------
#define AT_LD 72

__device__ __forceinline__ void flash_update(v8f* sf, float* mrun, float* lrun,
                                             v8f* o, _Float16* pdst) {
  const int lane = threadIdx.x & 31;
  const int ncol = lane & 15;
  const int half8 = (lane >> 4) << 3;
#pragma unroll
  for (int r = 0; r < 8; ++r) {
    float tmax = fmaxf(fmaxf(sf[0][r], sf[1][r]), fmaxf(sf[2][r], sf[3][r]));
#pragma unroll
    for (int off = 8; off >= 1; off >>= 1)
      tmax = fmaxf(tmax, __shfl_xor(tmax, off, 32));
    float newm = fmaxf(mrun[r], tmax);
    float alpha = __expf(mrun[r] - newm);
    mrun[r] = newm;
    float tsum = 0.0f;
#pragma unroll
    for (int sub = 0; sub < 4; ++sub) {
      float p = __expf(sf[sub][r] - newm);
      tsum += p;
      pdst[(r + half8) * AT_LD + sub * 16 + ncol] = (_Float16)p;
    }
#pragma unroll
    for (int off = 8; off >= 1; off >>= 1) tsum += __shfl_xor(tsum, off, 32);
    lrun[r] = alpha * lrun[r] + tsum;
#pragma unroll
    for (int nt = 0; nt < 4; ++nt) o[nt][r] *= alpha;
  }
}

__global__ __launch_bounds__(128) void diff_attn(const float* __restrict__ qproj,
                                                 const float* __restrict__ kproj,
                                                 const float* __restrict__ vproj,
                                                 const float* __restrict__ lam_p,
                                                 float* __restrict__ att) {
  __shared__ alignas(16) _Float16 k1s[64 * AT_LD];  // [key][d]
  __shared__ alignas(16) _Float16 k2s[64 * AT_LD];  // [key][d]
  __shared__ alignas(16) _Float16 vTs[64 * AT_LD];  // [d][key]
  __shared__ alignas(16) _Float16 p1s[64 * AT_LD];  // [qrow][key]
  __shared__ alignas(16) _Float16 p2s[64 * AT_LD];  // [qrow][key]

  const int t = threadIdx.x;
  const int wave = t >> 5;
  const int lane = t & 31;
  const int bh = blockIdx.x;
  const int qt = blockIdx.y;
  const int b = bh >> 3, h = bh & 7;

  // Persistent Q fragments: rows = qt*64 + wave*16 + m, cols: q1 d0..63, q2 d64..127
  const float* qbase =
      qproj + (((size_t)(b * SS + qt * 64 + wave * 16) * HH + h) * 192);
  const int qld = HH * 192;  // stride between consecutive s rows
  v16h a1[2], a2[2];
  a1[0] = load_A_gf32(qbase, qld, 0);
  a1[1] = load_A_gf32(qbase, qld, 32);
  a2[0] = load_A_gf32(qbase + 64, qld, 0);
  a2[1] = load_A_gf32(qbase + 64, qld, 32);

  v8f o1[4], o2[4];
#pragma unroll
  for (int i = 0; i < 4; ++i) { o1[i] = vzero8(); o2[i] = vzero8(); }
  float m1[8], l1[8], m2[8], l2[8];
#pragma unroll
  for (int r = 0; r < 8; ++r) {
    m1[r] = -1e30f; m2[r] = -1e30f; l1[r] = 0.0f; l2[r] = 0.0f;
  }

  const float inv = 0.125f;  // 1/sqrt(64)

  for (int ktile = 0; ktile < 16; ++ktile) {
    const int k0 = ktile * 64;
    __syncthreads();
    // stage K1,K2 ([key][d]) and V transposed ([d][key]) as f16.
    // Gather all global data first (8 guard-free reps), then convert+store.
    v4f kx1[8], kx2[8], vx[8];
#pragma unroll
    for (int rep = 0; rep < 8; ++rep) {
      int i = rep * 128 + t;
      int key = i >> 4, c4 = (i & 15) << 2;
      const float* kp =
          kproj + (((size_t)(b * SS + k0 + key) * HH + h) * 128) + c4;
      kx1[rep] = *(const v4f*)(kp);
      kx2[rep] = *(const v4f*)(kp + 64);
      const float* vp =
          vproj + (((size_t)(b * SS + k0 + key) * HH + h) * 64) + c4;
      vx[rep] = *(const v4f*)(vp);
    }
    // prefetch next key-tile while this one is processed
    if (ktile < 15) {
      int key = t >> 1, part = (t & 1) * 32;
      const float* kp =
          kproj + (((size_t)(b * SS + k0 + 64 + key) * HH + h) * 128) + part;
      __builtin_prefetch(kp, 0, 0);
      __builtin_prefetch(kp + 64, 0, 0);
      const float* vp =
          vproj + (((size_t)(b * SS + k0 + 64 + key) * HH + h) * 64) + part;
      __builtin_prefetch(vp, 0, 0);
    }
#pragma unroll
    for (int rep = 0; rep < 8; ++rep) {
      int i = rep * 128 + t;
      int key = i >> 4, c4 = (i & 15) << 2;
      _Float16* d1 = k1s + key * AT_LD + c4;
      _Float16* d2 = k2s + key * AT_LD + c4;
#pragma unroll
      for (int q = 0; q < 4; ++q) {
        d1[q] = (_Float16)kx1[rep][q];
        d2[q] = (_Float16)kx2[rep][q];
        vTs[(c4 + q) * AT_LD + key] = (_Float16)vx[rep][q];
      }
    }
    __syncthreads();

    v8f sf[4];
    // ---- scores 1 ----
#pragma unroll
    for (int sub = 0; sub < 4; ++sub) {
      v8f s = vzero8();
      s = wmma_f16(a1[0], load_B_lds(k1s, AT_LD, sub * 16, 0), s);
      s = wmma_f16(a1[1], load_B_lds(k1s, AT_LD, sub * 16, 32), s);
#pragma unroll
      for (int i = 0; i < 8; ++i) s[i] *= inv;
      sf[sub] = s;
    }
    flash_update(sf, m1, l1, o1, p1s + wave * 16 * AT_LD);
    // ---- scores 2 ----
#pragma unroll
    for (int sub = 0; sub < 4; ++sub) {
      v8f s = vzero8();
      s = wmma_f16(a2[0], load_B_lds(k2s, AT_LD, sub * 16, 0), s);
      s = wmma_f16(a2[1], load_B_lds(k2s, AT_LD, sub * 16, 32), s);
#pragma unroll
      for (int i = 0; i < 8; ++i) s[i] *= inv;
      sf[sub] = s;
    }
    flash_update(sf, m2, l2, o2, p2s + wave * 16 * AT_LD);

    // ---- P @ V ----
#pragma unroll
    for (int c = 0; c < 2; ++c) {
      v16h pa1 = load_A_lds(p1s, AT_LD, wave * 16, c * 32);
      v16h pa2 = load_A_lds(p2s, AT_LD, wave * 16, c * 32);
#pragma unroll
      for (int nt = 0; nt < 4; ++nt) {
        v16h bv = load_B_lds(vTs, AT_LD, nt * 16, c * 32);
        o1[nt] = wmma_f16(pa1, bv, o1[nt]);
        o2[nt] = wmma_f16(pa2, bv, o2[nt]);
      }
    }
  }

  // epilogue: out = O1/l1 - lam * O2/l2, layout att[b][h][s][d]
  const float lam = lam_p[0];
  const int ncol = lane & 15;
  const int qrow0 = qt * 64 + wave * 16 + ((lane >> 4) << 3);
#pragma unroll
  for (int nt = 0; nt < 4; ++nt) {
#pragma unroll
    for (int r = 0; r < 8; ++r) {
      float val = o1[nt][r] / l1[r] - lam * (o2[nt][r] / l2[r]);
      att[(((size_t)(b * HH + h) * SS) + qrow0 + r) * 64 + nt * 16 + ncol] = val;
    }
  }
}

// ---------------------------------------------------------------------------
// Kernel 3: group statistics per (b, j): mean/var over (h, s, e) 65536 values
// ---------------------------------------------------------------------------
__global__ __launch_bounds__(256) void group_stats(const float* __restrict__ att,
                                                   float* __restrict__ stats) {
  const int b = blockIdx.x >> 3, j = blockIdx.x & 7;
  float s = 0.0f, ss = 0.0f;
  for (int i = threadIdx.x; i < 16384; i += 256) {
    int h = i >> 11;
    int srow = (i >> 1) & 1023;
    int e4 = (i & 1) << 2;
    v4f v = *(const v4f*)(att +
                          (((size_t)(b * HH + h) * SS) + srow) * 64 + j * 8 + e4);
#pragma unroll
    for (int q = 0; q < 4; ++q) { s += v[q]; ss += v[q] * v[q]; }
  }
  __shared__ float red[512];
  red[threadIdx.x] = s;
  red[256 + threadIdx.x] = ss;
  __syncthreads();
  for (int st = 128; st > 0; st >>= 1) {
    if ((int)threadIdx.x < st) {
      red[threadIdx.x] += red[threadIdx.x + st];
      red[256 + threadIdx.x] += red[256 + threadIdx.x + st];
    }
    __syncthreads();
  }
  if (threadIdx.x == 0) {
    float mean = red[0] * (1.0f / 65536.0f);
    float var = red[256] * (1.0f / 65536.0f) - mean * mean;
    stats[blockIdx.x * 2] = mean;
    stats[blockIdx.x * 2 + 1] = rsqrtf(var + 0.001f);
  }
}

// ---------------------------------------------------------------------------
// Kernel 4: normalize + gamma/beta + (1-lambda_init) + sigmoid(gate)
// out flat index == ((b*S+s)*H + h)*64 + d  (matches (B,S,H*D))
// ---------------------------------------------------------------------------
__global__ __launch_bounds__(256) void finalize(const float* __restrict__ att,
                                                const float* __restrict__ qproj,
                                                const float* __restrict__ stats,
                                                const float* __restrict__ gamma,
                                                const float* __restrict__ beta,
                                                const float* __restrict__ li_p,
                                                float* __restrict__ out) {
  size_t idx = (size_t)blockIdx.x * 256 + threadIdx.x;
  if (idx >= (size_t)BB * SS * HH * DD) return;
  int d = idx & 63;
  size_t r = idx >> 6;
  int h = r & 7;
  size_t r2 = r >> 3;
  int srow = r2 & 1023;
  int b = (int)(r2 >> 10);
  float v = att[(((size_t)(b * HH + h) * SS) + srow) * 64 + d];
  int j = d >> 3;
  float mean = stats[(b * 8 + j) * 2];
  float istd = stats[(b * 8 + j) * 2 + 1];
  float g = qproj[(((size_t)(b * SS + srow) * HH) + h) * 192 + 128 + d];
  float val = (v - mean) * istd * gamma[d] + beta[d];
  val *= (1.0f - li_p[0]);
  val *= 1.0f / (1.0f + __expf(-g));
  out[idx] = val;
}

// ---------------------------------------------------------------------------
extern "C" void kernel_launch(void* const* d_in, const int* in_sizes, int n_in,
                              void* d_out, int out_size, void* d_ws,
                              size_t ws_size, hipStream_t stream) {
  (void)in_sizes; (void)n_in; (void)out_size; (void)ws_size;
  const float* query  = (const float*)d_in[0];
  const float* key    = (const float*)d_in[1];
  const float* values = (const float*)d_in[2];
  const float* Wq = (const float*)d_in[3];
  const float* bq = (const float*)d_in[4];
  const float* Wk = (const float*)d_in[5];
  const float* bk = (const float*)d_in[6];
  const float* Wv = (const float*)d_in[7];
  const float* bv = (const float*)d_in[8];
  const float* gamma = (const float*)d_in[9];
  const float* beta  = (const float*)d_in[10];
  const float* lam   = (const float*)d_in[11];
  const float* lam0  = (const float*)d_in[12];

  char* ws = (char*)d_ws;
  const size_t qproj_b = (size_t)BB * SS * 1536 * 4;  // 50331648
  const size_t kproj_b = (size_t)BB * SS * 1024 * 4;  // 33554432
  const size_t vproj_b = (size_t)BB * SS * 512 * 4;   // 16777216
  const size_t att_b   = (size_t)BB * HH * SS * DD * 4;
  float* qproj = (float*)ws;
  float* kproj = (float*)(ws + qproj_b);
  float* vproj = (float*)(ws + qproj_b + kproj_b);
  float* att   = (float*)(ws + qproj_b + kproj_b + vproj_b);
  float* stats = (float*)(ws + qproj_b + kproj_b + vproj_b + att_b);

  dim3 blk(128);
  gemm_proj<<<dim3(128, 24), blk, 0, stream>>>(query, Wq, bq, qproj, 8192, 1536, 512);
  gemm_proj<<<dim3(128, 16), blk, 0, stream>>>(key, Wk, bk, kproj, 8192, 1024, 512);
  gemm_proj<<<dim3(128, 8), blk, 0, stream>>>(values, Wv, bv, vproj, 8192, 512, 512);
  diff_attn<<<dim3(64, 16), blk, 0, stream>>>(qproj, kproj, vproj, lam, att);
  group_stats<<<64, 256, 0, stream>>>(att, stats);
  finalize<<<16384, 256, 0, stream>>>(att, qproj, stats, gamma, beta, lam0,
                                      (float*)d_out);
}
